// SimCLRLoss_79637283602639
// MI455X (gfx1250) — matrix operand order repaired and checked by reading, not compiled, not run on hardware
//
#include <hip/hip_runtime.h>
#include <math.h>

typedef __attribute__((ext_vector_type(16))) _Float16 v16h;
typedef __attribute__((ext_vector_type(8)))  float    v8f;

#define N_ROWS 8192
#define DIM    512
#define HALF_N 4096
#define WAVES  8
#define THREADS (WAVES * 32)
#define NTILES (N_ROWS / 16)          // 512 column tiles
#define LSTRIDE 520   // f16 elements per LDS row (512 + 8 pad -> +4 dword banks/row)

union Frag {
    uint4 q[2];
    v16h  h;
};

// ---------------------------------------------------------------------------
// Kernel 1: row L2-normalize, emit f16 in row-major (fnh) and transposed
// (fnhT, 512 x 8192) layouts. One block = 16 rows, 8 waves x 2 rows each.
// ---------------------------------------------------------------------------
__global__ void __launch_bounds__(256)
simclr_normalize(const float* __restrict__ feats,
                 _Float16* __restrict__ fnh,
                 _Float16* __restrict__ fnhT) {
    __shared__ _Float16 tile[16][LSTRIDE];

    const int i0   = blockIdx.x * 16;
    const int wave = threadIdx.x >> 5;
    const int lane = threadIdx.x & 31;

    #pragma unroll
    for (int rr = 0; rr < 2; ++rr) {
        const int row = wave * 2 + rr;
        const float4* src = (const float4*)(feats + (size_t)(i0 + row) * DIM);

        float4 v[4];
        float ss = 0.f;
        #pragma unroll
        for (int c = 0; c < 4; ++c) {
            v[c] = src[c * 32 + lane];
            ss += v[c].x * v[c].x + v[c].y * v[c].y + v[c].z * v[c].z + v[c].w * v[c].w;
        }
        #pragma unroll
        for (int m = 16; m >= 1; m >>= 1) ss += __shfl_xor(ss, m, 32);

        const float scale = 1.0f / fmaxf(sqrtf(ss), 1e-8f);

        #pragma unroll
        for (int c = 0; c < 4; ++c) {
            const int k = (c * 32 + lane) * 4;
            alignas(8) _Float16 h4[4];
            h4[0] = (_Float16)(v[c].x * scale);
            h4[1] = (_Float16)(v[c].y * scale);
            h4[2] = (_Float16)(v[c].z * scale);
            h4[3] = (_Float16)(v[c].w * scale);
            *(uint2*)(fnh + (size_t)(i0 + row) * DIM + k) = *(const uint2*)h4;
            *(uint2*)(&tile[row][k])                      = *(const uint2*)h4;
        }
    }
    __syncthreads();

    // transposed writes: fnhT[k][i0..i0+15], 32 contiguous bytes per k
    for (int k = threadIdx.x; k < DIM; k += 256) {
        alignas(16) _Float16 tmp[16];
        #pragma unroll
        for (int r = 0; r < 16; ++r) tmp[r] = tile[r][k];
        uint4* dst = (uint4*)(fnhT + (size_t)k * N_ROWS + i0);
        dst[0] = ((const uint4*)tmp)[0];
        dst[1] = ((const uint4*)tmp)[1];
    }
}

// ---------------------------------------------------------------------------
// Kernel 2: fused cos-sim GEMM (WMMA f16 -> f32) + masked logsumexp + pos pair.
// One block = one 16-row strip; 8 waves, each computing 2 adjacent N-tiles per
// A-fragment, sweeping all 512 column tiles.
// ---------------------------------------------------------------------------
__global__ void __launch_bounds__(THREADS)
simclr_main(const _Float16* __restrict__ fnh,
            const _Float16* __restrict__ fnhT,
            float* __restrict__ partials) {
    __shared__ _Float16 strip[16][LSTRIDE];
    __shared__ float rowsum[16];
    __shared__ float posv[16];

    const int i0  = blockIdx.x * 16;
    const int tid = threadIdx.x;

    // stage A strip (16 x 512 f16) into LDS, coalesced
    {
        const int row = tid >> 4;
        const int seg = tid & 15;                 // 32 f16 per segment
        const uint4* src = (const uint4*)(fnh + (size_t)(i0 + row) * DIM + seg * 32);
        uint4* dst = (uint4*)(&strip[row][seg * 32]);
        #pragma unroll
        for (int c = 0; c < 4; ++c) dst[c] = src[c];
    }
    if (tid < 16) { rowsum[tid] = 0.f; posv[tid] = 0.f; }
    __syncthreads();

    const int wave = tid >> 5;
    const int lane = tid & 31;
    const int hsel = lane >> 4;       // 0: lanes 0-15, 1: lanes 16-31
    const int l16  = lane & 15;

    const int diag_jt = i0 >> 4;                               // tile holding j == i
    const int pos_jt  = ((i0 + HALF_N) & (N_ROWS - 1)) >> 4;   // tile holding j == (i+N/2)%N

    float lacc[8];
    #pragma unroll
    for (int r = 0; r < 8; ++r) lacc[r] = 0.f;
    float lpos = 0.f;

    // branchless-fast-path tile epilogue (uniform scalar branch per tile)
    auto process = [&](const v8f& c, int jt) {
        const bool sd = (jt == diag_jt);
        const bool sp = (jt == pos_jt);
        if (!sd && !sp) {
            #pragma unroll
            for (int r = 0; r < 8; ++r) lacc[r] += __expf(c[r]);
        } else {
            #pragma unroll
            for (int r = 0; r < 8; ++r) {
                const bool mine = (l16 == hsel * 8 + r);   // element on row i0+l16
                if (!(sd && mine)) lacc[r] += __expf(c[r]);
                if (sp && mine)    lpos += c[r];
            }
        }
    };

    for (int t = 0; t < NTILES / (WAVES * 2); ++t) {
        const int jt0   = t * (WAVES * 2) + wave * 2;          // two adjacent tiles
        const int jbase = jt0 * 16;
        v8f c0 = {}, c1 = {};

        #pragma unroll 2
        for (int kb = 0; kb < DIM / 32; ++kb) {
            const int k0 = kb * 32;
            Frag a, b0, b1;
            // A fragment (16x32 f16): lane m holds K = {hsel*8+0..7, 16+hsel*8+0..7}
            a.q[0] = *(const uint4*)(&strip[l16][k0 + hsel * 8]);
            a.q[1] = *(const uint4*)(&strip[l16][k0 + 16 + hsel * 8]);
            // B fragments (32x16 f16 each): lane n holds K = {hsel*8+0..7, 16+hsel*8+0..7}
            // of column jbase+n (b0) / jbase+16+n (b1), read from row-major fnh
            const _Float16* bp0 = fnh + (size_t)(jbase + l16) * DIM + k0 + hsel * 8;
            const _Float16* bp1 = bp0 + (size_t)16 * DIM;
            b0.q[0] = *(const uint4*)(bp0);
            b0.q[1] = *(const uint4*)(bp0 + 16);
            b1.q[0] = *(const uint4*)(bp1);
            b1.q[1] = *(const uint4*)(bp1 + 16);
            c0 = __builtin_amdgcn_wmma_f32_16x16x32_f16(
                     false, a.h, false, b0.h, (short)0, c0, false, false);
            c1 = __builtin_amdgcn_wmma_f32_16x16x32_f16(
                     false, a.h, false, b1.h, (short)0, c1, false, false);
        }

        process(c0, jt0);
        process(c1, jt0 + 1);
    }

    // reduce per-row exp-sums across the 16 column-lanes of each half
    #pragma unroll
    for (int r = 0; r < 8; ++r) {
        float s = lacc[r];
        #pragma unroll
        for (int m = 8; m >= 1; m >>= 1) s += __shfl_xor(s, m, 16);
        if (l16 == 0) atomicAdd(&rowsum[hsel * 8 + r], s);
    }
    // lane's lpos (if any) belongs to local row l16; zero elsewhere
    atomicAdd(&posv[l16], lpos);
    __syncthreads();

    if (tid < 16) {
        float v = __logf(rowsum[tid]) - posv[tid];
        #pragma unroll
        for (int m = 8; m >= 1; m >>= 1) v += __shfl_xor(v, m, 16);
        if (tid == 0) partials[blockIdx.x] = v;
    }
}

// ---------------------------------------------------------------------------
// Kernel 3: sum 512 per-strip partials into the scalar output.
// ---------------------------------------------------------------------------
__global__ void __launch_bounds__(256)
simclr_reduce(const float* __restrict__ partials, float* __restrict__ out, int n) {
    __shared__ float sdata[256];
    float s = 0.f;
    for (int i = threadIdx.x; i < n; i += 256) s += partials[i];
    sdata[threadIdx.x] = s;
    __syncthreads();
    #pragma unroll
    for (int m = 128; m >= 1; m >>= 1) {
        if (threadIdx.x < m) sdata[threadIdx.x] += sdata[threadIdx.x + m];
        __syncthreads();
    }
    if (threadIdx.x == 0) out[0] = sdata[0];
}

extern "C" void kernel_launch(void* const* d_in, const int* in_sizes, int n_in,
                              void* d_out, int out_size, void* d_ws, size_t ws_size,
                              hipStream_t stream) {
    (void)in_sizes; (void)n_in; (void)out_size; (void)ws_size;

    const float* feats = (const float*)d_in[0];
    char* ws = (char*)d_ws;

    _Float16* fnh  = (_Float16*)ws;                                     // 8 MB
    _Float16* fnhT = (_Float16*)(ws + (size_t)N_ROWS * DIM * 2);        // 8 MB
    float* partials = (float*)(ws + (size_t)N_ROWS * DIM * 4);          // 2 KB

    simclr_normalize<<<N_ROWS / 16, 256, 0, stream>>>(feats, fnh, fnhT);
    simclr_main<<<N_ROWS / 16, THREADS, 0, stream>>>(fnh, fnhT, partials);
    simclr_reduce<<<1, 256, 0, stream>>>(partials, (float*)d_out, N_ROWS / 16);
}
